// GLSTM7_55078660604357
// MI455X (gfx1250) — matrix operand order repaired and checked by reading, not compiled, not run on hardware
//
#include <hip/hip_runtime.h>
#include <hip/hip_bf16.h>
#include <math.h>

#define N_NODES 10000
#define EDGES   320000
#define DIN     128
#define HDIM    256
#define GDIM    1024   // 4*H

typedef __attribute__((ext_vector_type(16))) __bf16 v16bf;
typedef __attribute__((ext_vector_type(8)))  float  v8f;

// ---------- helpers ----------
static __device__ __forceinline__ unsigned short f2bf(float f) {
    union { float f; unsigned u; } x; x.f = f;
    unsigned u = x.u;
    u += 0x7FFFu + ((u >> 16) & 1u);          // round to nearest even
    return (unsigned short)(u >> 16);
}
static __device__ __forceinline__ float sigf(float x) {
    return 1.0f / (1.0f + __expf(-x));
}

// ---------- small elementwise kernels ----------
__global__ void k_f32_to_bf16(const float* __restrict__ in,
                              unsigned short* __restrict__ out, int n) {
    int i = blockIdx.x * blockDim.x + threadIdx.x;
    if (i < n) out[i] = f2bf(in[i]);
}

// in: [K,N] f32 row-major  ->  out: [N,K] bf16 row-major (B-transposed layout)
__global__ void k_transpose_bf16(const float* __restrict__ in,
                                 unsigned short* __restrict__ out, int K, int N) {
    int i = blockIdx.x * blockDim.x + threadIdx.x;
    if (i < K * N) {
        int k = i / N, n = i % N;
        out[n * K + k] = f2bf(in[i]);
    }
}

__global__ void k_zero(float* __restrict__ p, int n) {
    int i = blockIdx.x * blockDim.x + threadIdx.x;
    if (i < n) p[i] = 0.0f;
}

__global__ void k_addvec(const float* __restrict__ a, const float* __restrict__ b,
                         float* __restrict__ o, int n) {
    int i = blockIdx.x * blockDim.x + threadIdx.x;
    if (i < n) o[i] = a[i] + b[i];
}

// ---------- graph degree / normalization ----------
__global__ void k_deg(const long long* __restrict__ dst, float* __restrict__ deg, int E) {
    int e = blockIdx.x * blockDim.x + threadIdx.x;
    if (e < E) atomicAdd(&deg[(int)dst[e]], 1.0f);
}
__global__ void k_dinv(float* __restrict__ d, int n) {   // in place: dinv = rsqrt(deg+1)
    int i = blockIdx.x * blockDim.x + threadIdx.x;
    if (i < n) d[i] = rsqrtf(d[i] + 1.0f);
}

// ---------- GCN scatter-add:  agg[dst] += dinv[src]*dinv[dst]*xw[src] ----------
__global__ void k_scatter(const long long* __restrict__ src, const long long* __restrict__ dst,
                          const float* __restrict__ dinv, const float* __restrict__ xw,
                          float* __restrict__ agg, int E) {
    int idx = blockIdx.x * blockDim.x + threadIdx.x;   // E * 64 threads (4 feats each)
    int e = idx >> 6;
    if (e >= E) return;
    int f = (idx & 63) << 2;
    int s = (int)src[e], d = (int)dst[e];
    float w = dinv[s] * dinv[d];
    float4 v = *(const float4*)(xw + (size_t)s * HDIM + f);
    float* o = agg + (size_t)d * HDIM + f;
    atomicAdd(o + 0, w * v.x);
    atomicAdd(o + 1, w * v.y);
    atomicAdd(o + 2, w * v.z);
    atomicAdd(o + 3, w * v.w);
}

// out = relu(agg + dinv^2 * xw + b)
__global__ void k_finish_conv(const float* __restrict__ agg, const float* __restrict__ xw,
                              const float* __restrict__ dinv, const float* __restrict__ b,
                              float* __restrict__ out, int total) {
    int i = blockIdx.x * blockDim.x + threadIdx.x;
    if (i >= total) return;
    int node = i / HDIM, f = i % HDIM;
    float di = dinv[node];
    float v = agg[i] + di * di * xw[i] + b[f];
    out[i] = v > 0.0f ? v : 0.0f;
}

// ---------- WMMA bf16 GEMM:  C[M,N] = A[M,K] * Bt[N,K]^T (+bias) ----------
// Each wave computes a 16x64 macro-tile: the A fragment is loaded once per
// K-chunk and reused across 4 WMMAs (4x arithmetic intensity on the A path).
//
// A fragment (16x32 bf16): lane holds row lane&15; lane-half h picks K slots
// {8h..8h+7, 16+8h..16+8h+7}  -> two contiguous 16B loads.
static __device__ __forceinline__ v16bf load_frag_a(const unsigned short* __restrict__ base,
                                                    int ld, int k0, int lane) {
    int r = lane & 15, h = lane >> 4;
    const unsigned short* p = base + (size_t)r * ld + k0 + 8 * h;
    union { v16bf v; uint4 q[2]; } u;
    u.q[0] = *(const uint4*)p;
    u.q[1] = *(const uint4*)(p + 16);
    return u.v;
}
// B fragment (32x16 bf16): lane holds column lane&15; lane-half h picks K slots
// {8h..8h+7, 16+8h..16+8h+7}  -> two contiguous 16B loads.
static __device__ __forceinline__ v16bf load_frag_b(const unsigned short* __restrict__ base,
                                                    int ld, int k0, int lane) {
    int n = lane & 15, h = lane >> 4;
    const unsigned short* p = base + (size_t)n * ld + k0 + 8 * h;
    union { v16bf v; uint4 q[2]; } u;
    u.q[0] = *(const uint4*)p;
    u.q[1] = *(const uint4*)(p + 16);
    return u.v;
}

__global__ __launch_bounds__(256) void k_gemm_bf16(
    const unsigned short* __restrict__ A,   // [M,K] bf16
    const unsigned short* __restrict__ Bt,  // [N,K] bf16 (B transposed)
    float* __restrict__ C,                  // [M,N] f32
    const float* __restrict__ bias,         // [N] or nullptr
    int M, int N, int K) {
    int lane    = threadIdx.x & 31;
    int wave    = blockIdx.x * (blockDim.x >> 5) + (threadIdx.x >> 5);
    int macrosN = N >> 6;                    // 64-wide macro-columns
    int macros  = (M >> 4) * macrosN;
    if (wave >= macros) return;
    int tm = wave / macrosN, tn = wave % macrosN;   // tn fastest: adjacent waves share A rows
    const unsigned short* a0 = A + (size_t)(tm * 16) * K;
    const unsigned short* b0 = Bt + (size_t)(tn * 64) * K;
    const size_t bstride = (size_t)16 * K;

    v8f acc0 = {}, acc1 = {}, acc2 = {}, acc3 = {};
    for (int k0 = 0; k0 < K; k0 += 32) {
        v16bf af  = load_frag_a(a0, K, k0, lane);
        v16bf bf0 = load_frag_b(b0,               K, k0, lane);
        v16bf bf1 = load_frag_b(b0 + bstride,     K, k0, lane);
        v16bf bf2 = load_frag_b(b0 + 2 * bstride, K, k0, lane);
        v16bf bf3 = load_frag_b(b0 + 3 * bstride, K, k0, lane);
        acc0 = __builtin_amdgcn_wmma_f32_16x16x32_bf16(false, af, false, bf0, (short)0, acc0, false, false);
        acc1 = __builtin_amdgcn_wmma_f32_16x16x32_bf16(false, af, false, bf1, (short)0, acc1, false, false);
        acc2 = __builtin_amdgcn_wmma_f32_16x16x32_bf16(false, af, false, bf2, (short)0, acc2, false, false);
        acc3 = __builtin_amdgcn_wmma_f32_16x16x32_bf16(false, af, false, bf3, (short)0, acc3, false, false);
    }
    int nl = lane & 15;
    int mh = (lane >> 4) * 8;
    v8f accs[4] = { acc0, acc1, acc2, acc3 };
    #pragma unroll
    for (int i = 0; i < 4; ++i) {
        int n = tn * 64 + i * 16 + nl;
        float bb = bias ? bias[n] : 0.0f;
        #pragma unroll
        for (int v = 0; v < 8; ++v) {
            int m = tm * 16 + mh + v;
            C[(size_t)m * N + n] = accs[i][v] + bb;
        }
    }
}

// ---------- sequential LSTM scan (recurrent GEMV only; input proj hoisted) ----------
__global__ __launch_bounds__(1024) void k_lstm_seq(
    const float* __restrict__ G,     // [T, 1024] precomputed x@Wih.T + bih + bhh
    const float* __restrict__ Whh,   // [1024, 256] f32 (L2-resident)
    float* __restrict__ Hout,        // [T, 256]
    int T) {
    __shared__ __align__(16) float h[HDIM];
    __shared__ __align__(16) float c[HDIM];
    __shared__ __align__(16) float gsh[GDIM];
    int j = threadIdx.x;               // one thread per gate row
    if (j < HDIM) { h[j] = 0.0f; c[j] = 0.0f; }
    __syncthreads();
    const float4* wr = (const float4*)(Whh + (size_t)j * HDIM);
    for (int t = 0; t < T; ++t) {
        if (t + 1 < T) __builtin_prefetch(G + (size_t)(t + 1) * GDIM + j, 0, 1);
        float acc = G[(size_t)t * GDIM + j];
        #pragma unroll 8
        for (int k = 0; k < HDIM / 4; ++k) {
            float4 w  = wr[k];
            float4 hv = ((const float4*)h)[k];
            acc += w.x * hv.x + w.y * hv.y + w.z * hv.z + w.w * hv.w;
        }
        gsh[j] = acc;
        __syncthreads();
        if (j < HDIM) {
            float ig = sigf(gsh[j]);
            float fg = sigf(gsh[j + HDIM]);
            float gg = tanhf(gsh[j + 2 * HDIM]);
            float og = sigf(gsh[j + 3 * HDIM]);
            float cn = fg * c[j] + ig * gg;
            float hn = og * tanhf(cn);
            c[j] = cn; h[j] = hn;
            Hout[(size_t)t * HDIM + j] = hn;
        }
        __syncthreads();
    }
}

// ---------- final FC + sigmoid ----------
__global__ void k_fc_sigmoid(const float* __restrict__ Hf, const float* __restrict__ Wfc,
                             const float* __restrict__ bfc, float* __restrict__ out, int n) {
    int i = blockIdx.x * blockDim.x + threadIdx.x;
    if (i >= n) return;
    const float4* hp = (const float4*)(Hf + (size_t)i * HDIM);
    const float4* wp = (const float4*)Wfc;
    float acc = bfc[0];
    for (int k = 0; k < HDIM / 4; ++k) {
        float4 a = hp[k], b = wp[k];
        acc += a.x * b.x + a.y * b.y + a.z * b.z + a.w * b.w;
    }
    out[i] = sigf(acc);
}

// ---------- host orchestration ----------
static inline int nblk(long long n, int b) { return (int)((n + b - 1) / b); }

extern "C" void kernel_launch(void* const* d_in, const int* in_sizes, int n_in,
                              void* d_out, int out_size, void* d_ws, size_t ws_size,
                              hipStream_t stream) {
    (void)in_sizes; (void)n_in; (void)out_size; (void)ws_size;
    const float*     x    = (const float*)d_in[0];
    const long long* ei   = (const long long*)d_in[1];
    const long long* srcI = ei;
    const long long* dstI = ei + EDGES;
    const float* W[5]  = { (const float*)d_in[2], (const float*)d_in[4], (const float*)d_in[6],
                           (const float*)d_in[8], (const float*)d_in[10] };
    const float* B[5]  = { (const float*)d_in[3], (const float*)d_in[5], (const float*)d_in[7],
                           (const float*)d_in[9], (const float*)d_in[11] };
    const float* Wih0 = (const float*)d_in[12]; const float* Whh0 = (const float*)d_in[13];
    const float* bih0 = (const float*)d_in[14]; const float* bhh0 = (const float*)d_in[15];
    const float* Wih1 = (const float*)d_in[16]; const float* Whh1 = (const float*)d_in[17];
    const float* bih1 = (const float*)d_in[18]; const float* bhh1 = (const float*)d_in[19];
    const float* Wfc  = (const float*)d_in[20]; const float* bfc  = (const float*)d_in[21];

    // workspace layout (256B aligned slices)
    char* base = (char*)d_ws;
    size_t off = 0;
    auto alloc = [&](size_t bytes) { size_t o = off; off = (off + bytes + 255) & ~(size_t)255; return o; };
    float* bufA = (float*)(base + alloc((size_t)N_NODES * HDIM * 4));
    float* bufB = (float*)(base + alloc((size_t)N_NODES * HDIM * 4));
    float* xw   = (float*)(base + alloc((size_t)N_NODES * HDIM * 4));
    float* agg  = (float*)(base + alloc((size_t)N_NODES * HDIM * 4));
    float* Gbuf = (float*)(base + alloc((size_t)N_NODES * GDIM * 4));
    float* dinv = (float*)(base + alloc((size_t)N_NODES * 4));
    float* bs0  = (float*)(base + alloc((size_t)GDIM * 4));
    float* bs1  = (float*)(base + alloc((size_t)GDIM * 4));
    unsigned short* actbf = (unsigned short*)(base + alloc((size_t)N_NODES * HDIM * 2));
    unsigned short* W1t   = (unsigned short*)(base + alloc((size_t)HDIM * DIN * 2));
    unsigned short* Wt[4];
    for (int i = 0; i < 4; ++i) Wt[i] = (unsigned short*)(base + alloc((size_t)HDIM * HDIM * 2));
    unsigned short* Wih0b = (unsigned short*)(base + alloc((size_t)GDIM * HDIM * 2));
    unsigned short* Wih1b = (unsigned short*)(base + alloc((size_t)GDIM * HDIM * 2));

    const int TB = 256;
    // --- weight prep (bf16, B-transposed layouts) ---
    k_transpose_bf16<<<nblk((long long)DIN * HDIM, TB), TB, 0, stream>>>(W[0], W1t, DIN, HDIM);
    for (int i = 1; i < 5; ++i)
        k_transpose_bf16<<<nblk((long long)HDIM * HDIM, TB), TB, 0, stream>>>(W[i], Wt[i - 1], HDIM, HDIM);
    k_f32_to_bf16<<<nblk((long long)GDIM * HDIM, TB), TB, 0, stream>>>(Wih0, Wih0b, GDIM * HDIM);
    k_f32_to_bf16<<<nblk((long long)GDIM * HDIM, TB), TB, 0, stream>>>(Wih1, Wih1b, GDIM * HDIM);
    k_addvec<<<nblk(GDIM, TB), TB, 0, stream>>>(bih0, bhh0, bs0, GDIM);
    k_addvec<<<nblk(GDIM, TB), TB, 0, stream>>>(bih1, bhh1, bs1, GDIM);

    // --- symmetric-norm coefficients ---
    k_zero<<<nblk(N_NODES, TB), TB, 0, stream>>>(dinv, N_NODES);
    k_deg<<<nblk(EDGES, TB), TB, 0, stream>>>(dstI, dinv, EDGES);
    k_dinv<<<nblk(N_NODES, TB), TB, 0, stream>>>(dinv, N_NODES);

    auto gemm = [&](const unsigned short* A, const unsigned short* Bt, float* C,
                    const float* bias, int M, int N, int K) {
        long long macros = (long long)(M / 16) * (N / 64);   // 16x64 macro-tiles per wave
        k_gemm_bf16<<<nblk(macros, 8), 256, 0, stream>>>(A, Bt, C, bias, M, N, K);
    };
    auto conv = [&](const float* hin, int K, const unsigned short* Wtb, const float* bias, float* hout) {
        k_f32_to_bf16<<<nblk((long long)N_NODES * K, TB), TB, 0, stream>>>(hin, actbf, N_NODES * K);
        gemm(actbf, Wtb, xw, nullptr, N_NODES, HDIM, K);
        k_zero<<<nblk((long long)N_NODES * HDIM, TB), TB, 0, stream>>>(agg, N_NODES * HDIM);
        k_scatter<<<nblk((long long)EDGES * 64, TB), TB, 0, stream>>>(srcI, dstI, dinv, xw, agg, EDGES);
        k_finish_conv<<<nblk((long long)N_NODES * HDIM, TB), TB, 0, stream>>>(agg, xw, dinv, bias, hout,
                                                                              N_NODES * HDIM);
    };

    // conv1: x[10000,128] -> bufA
    conv(x, DIN, W1t, B[0], bufA);

    // LSTM layer 0: hoisted input GEMM + sequential scan -> bufB
    k_f32_to_bf16<<<nblk((long long)N_NODES * HDIM, TB), TB, 0, stream>>>(bufA, actbf, N_NODES * HDIM);
    gemm(actbf, Wih0b, Gbuf, bs0, N_NODES, GDIM, HDIM);
    k_lstm_seq<<<1, 1024, 0, stream>>>(Gbuf, Whh0, bufB, N_NODES);

    // LSTM layer 1 -> bufA
    k_f32_to_bf16<<<nblk((long long)N_NODES * HDIM, TB), TB, 0, stream>>>(bufB, actbf, N_NODES * HDIM);
    gemm(actbf, Wih1b, Gbuf, bs1, N_NODES, GDIM, HDIM);
    k_lstm_seq<<<1, 1024, 0, stream>>>(Gbuf, Whh1, bufA, N_NODES);

    // conv2..conv5 (ping-pong)
    conv(bufA, HDIM, Wt[0], B[1], bufB);
    conv(bufB, HDIM, Wt[1], B[2], bufA);
    conv(bufA, HDIM, Wt[2], B[3], bufB);
    conv(bufB, HDIM, Wt[3], B[4], bufA);

    // final FC + sigmoid -> d_out [10000]
    k_fc_sigmoid<<<nblk(N_NODES, TB), TB, 0, stream>>>(bufA, Wfc, bfc, (float*)d_out, N_NODES);
}
